// WavenetGenerator_57930518888906
// MI455X (gfx1250) — matrix operand order repaired and checked by reading, not compiled, not run on hardware
//
#include <hip/hip_runtime.h>
#include <hip/hip_bf16.h>

typedef _Float16 f16;
typedef __attribute__((ext_vector_type(8)))  _Float16 v8h;
typedef __attribute__((ext_vector_type(16))) _Float16 v16h;
typedef __attribute__((ext_vector_type(8)))  float    v8f;

// ---------------- problem constants ----------------
#define BATCH      4
#define N_ENC      4
#define CREP       200
#define TSTEPS     800      // N_ENC * CREP
#define LAYERS     16
#define RCH        128      // R
#define SCH        128      // S
#define CCH        64       // CC
#define CLASSES    256
#define NTHREADS   512      // 16 wave32 waves on one WGP

// packed-weight region sizes (in f16 elements)
// GEMM1: per layer K=320 (q0|res|c), N=256 -> KT=10, NT=16
#define P1_LAYER   (10*16*512)          // 81920
#define P1_TOTAL   (LAYERS*P1_LAYER)    // 1310720
// GEMM2: per layer K=128 (h), N=256 (skip|res) -> KT=4, NT=16
#define P2_LAYER   (4*16*512)           // 32768
#define P2_TOTAL   (LAYERS*P2_LAYER)    // 524288
// skip0: K=128, N=128 -> KT=4, NT=8
#define PS0_TOTAL  (4*8*512)            // 16384
// top fc: K=192 (relu(skip)|c), N=128 -> KT=6, NT=8
#define P3_TOTAL   (6*8*512)            // 24576
// logits: K=128, N=256 -> KT=4, NT=16
#define P4_TOTAL   (4*16*512)           // 32768
#define PACK_TOTAL (P1_TOTAL+P2_TOTAL+PS0_TOTAL+P3_TOTAL+P4_TOTAL) // 1908736

#define QFLOATS    (BATCH*RCH*510)      // sum of dilations = 510 -> 261120 floats

// workspace byte offsets (256B aligned)
#define WS_P1   0u
#define WS_P2   2621440u
#define WS_PS0  3670016u
#define WS_P3   3702784u
#define WS_P4   3751936u
#define WS_Q    3817472u

// ---------------- WMMA helpers ----------------
__device__ __forceinline__ v8f wmma16(v16h a, v16h b, v8f c) {
  return __builtin_amdgcn_wmma_f32_16x16x32_f16(false, a, false, b, (short)0, c,
                                                false, false);
}

// A fragment (16x32 f16) from LDS, row-major [16][strideH]
// lane layout: M = lane%16 ; VGPR0..7 K=kb..kb+15,
// kb = kt*32 + (lane<16 ? 0 : 16)
__device__ __forceinline__ v16h ldsA(const f16* A, int strideH, int kt, int lane) {
  int m  = lane & 15;
  int kb = kt * 32 + ((lane & 16) ? 16 : 0);
  const f16* p = A + m * strideH + kb;
  v8h lo = *(const v8h*)(p);
  v8h hi = *(const v8h*)(p + 8);
  return __builtin_shufflevector(lo, hi, 0,1,2,3,4,5,6,7,8,9,10,11,12,13,14,15);
}

// B fragment (32x16 f16) from global, pre-packed per-lane contiguous:
// packed[tile*512 + lane*16 + j] = B[kt*32 + (lane<16?0:16) + j][nt*16 + lane%16]
__device__ __forceinline__ v16h glbB(const f16* B, int tile, int lane) {
  const f16* p = B + (size_t)tile * 512 + lane * 16;
  v8h lo = *(const v8h*)(p);
  v8h hi = *(const v8h*)(p + 8);
  return __builtin_shufflevector(lo, hi, 0,1,2,3,4,5,6,7,8,9,10,11,12,13,14,15);
}

// ---------------- weight pack / init kernel ----------------
__global__ void wavenet_pack(const float* __restrict__ Wc,
                             const float* __restrict__ Wcond,
                             const float* __restrict__ Wres,
                             const float* __restrict__ Wskip_l,
                             const float* __restrict__ Wskip0,
                             const float* __restrict__ Wcond_top,
                             const float* __restrict__ Wfc,
                             const float* __restrict__ Wlogits,
                             f16* __restrict__ P1, f16* __restrict__ P2,
                             f16* __restrict__ PS0, f16* __restrict__ P3,
                             f16* __restrict__ P4, float* __restrict__ Q)
{
  int p = blockIdx.x * blockDim.x + threadIdx.x;
  if (p < QFLOATS) Q[p] = 0.f;                 // zero dilation ring buffers
  if (p >= PACK_TOTAL) return;

  if (p < P1_TOTAL) {                          // gated conv + cond, K=320 N=256
    int l = p / P1_LAYER, e = p % P1_LAYER;
    int j = e & 15, lane = (e >> 4) & 31, tile = e >> 9;
    int kt = tile % 10, nt = tile / 10;
    int n = nt * 16 + (lane & 15);
    int k = kt * 32 + ((lane & 16) ? 16 : 0) + j;
    float v;
    if (k < 128)      v = Wc[(((l*256 + n)*128 + k) << 1) + 0];
    else if (k < 256) v = Wc[(((l*256 + n)*128 + (k-128)) << 1) + 1];
    else              v = Wcond[(l*256 + n)*64 + (k - 256)];
    P1[p] = (f16)v;
    return;
  }
  p -= P1_TOTAL;
  if (p < P2_TOTAL) {                          // skip|res, K=128 N=256
    int l = p / P2_LAYER, e = p % P2_LAYER;
    int j = e & 15, lane = (e >> 4) & 31, tile = e >> 9;
    int kt = tile % 4, nt = tile / 4;
    int n = nt * 16 + (lane & 15);
    int k = kt * 32 + ((lane & 16) ? 16 : 0) + j;
    float v = (n < 128) ? Wskip_l[(l*128 + n)*128 + k]
                        : Wres  [(l*128 + (n-128))*128 + k];
    P2[p - 0] = (f16)v;
    return;
  }
  p -= P2_TOTAL;
  if (p < PS0_TOTAL) {                         // skip0, K=128 N=128
    int j = p & 15, lane = (p >> 4) & 31, tile = p >> 9;
    int kt = tile % 4, nt = tile / 4;
    int n = nt * 16 + (lane & 15);
    int k = kt * 32 + ((lane & 16) ? 16 : 0) + j;
    PS0[p] = (f16)Wskip0[n*128 + k];
    return;
  }
  p -= PS0_TOTAL;
  if (p < P3_TOTAL) {                          // fc + cond_top, K=192 N=128
    int j = p & 15, lane = (p >> 4) & 31, tile = p >> 9;
    int kt = tile % 6, nt = tile / 6;
    int n = nt * 16 + (lane & 15);
    int k = kt * 32 + ((lane & 16) ? 16 : 0) + j;
    float v = (k < 128) ? Wfc[n*128 + k] : Wcond_top[n*64 + (k-128)];
    P3[p] = (f16)v;
    return;
  }
  p -= P3_TOTAL;
  {                                            // logits, K=128 N=256
    int j = p & 15, lane = (p >> 4) & 31, tile = p >> 9;
    int kt = tile % 4, nt = tile / 4;
    int n = nt * 16 + (lane & 15);
    int k = kt * 32 + ((lane & 16) ? 16 : 0) + j;
    P4[p] = (f16)Wlogits[n*128 + k];
  }
}

// ---------------- persistent single-WGP generation kernel ----------------
__launch_bounds__(NTHREADS, 1)
__global__ void wavenet_gen(const float* __restrict__ enc,
                            const float* __restrict__ Wfirst,
                            const float* __restrict__ bfirst,
                            const float* __restrict__ bc,
                            const float* __restrict__ bres,
                            const float* __restrict__ bskip_l,
                            const float* __restrict__ bskip0,
                            const float* __restrict__ bfc,
                            const float* __restrict__ blogits,
                            const f16* __restrict__ P1, const f16* __restrict__ P2,
                            const f16* __restrict__ PS0, const f16* __restrict__ P3,
                            const f16* __restrict__ P4,
                            float* __restrict__ Q, float* __restrict__ out)
{
  __shared__ __align__(16) f16 A1[16*320];   // [q0 | res | c_t] f16, B padded to 16
  __shared__ __align__(16) f16 A2[16*128];   // res / h / relu(s)  (rows 4..15 stay 0)
  __shared__ __align__(16) f16 A3[16*192];   // [relu(skip) | c_t]
  __shared__ float zbuf[16*256];             // z scratch (fp32)
  __shared__ float logitsS[4*256];
  __shared__ float resS[4*128];
  __shared__ float skipS[4*128];
  __shared__ float xprevS[4];
  __shared__ int   sampleS[4];

  const int tid  = threadIdx.x;
  const int lane = tid & 31;
  const int w    = tid >> 5;                 // wave id 0..15

  for (int i = tid; i < 16*320; i += NTHREADS) A1[i] = (f16)0.f;
  for (int i = tid; i < 16*128; i += NTHREADS) A2[i] = (f16)0.f;
  for (int i = tid; i < 16*192; i += NTHREADS) A3[i] = (f16)0.f;
  if (tid < 4) { sampleS[tid] = 128; xprevS[tid] = 0.f; }
  __syncthreads();

  for (int t = 0; t < TSTEPS; ++t) {
    // ---- S1: first causal conv (k=2), layer-0 queue r/w + A1 staging,
    //          conditioning frame load ----
    {
      int m = tid >> 7, r = tid & 127;
      float x  = (float)sampleS[m] * (1.f/127.5f) - 1.f;
      float fq = xprevS[m];
      float rv = fq * Wfirst[2*r+0] + x * Wfirst[2*r+1] + bfirst[r];
      resS[tid]     = rv;
      A2[m*128 + r] = (f16)rv;               // A for skip0 GEMM
      float* qp = Q + tid;                   // layer 0: d=1, slot=0
      float q0 = *qp;
      *qp = rv;
      A1[m*320 + r]       = (f16)q0;
      A1[m*320 + 128 + r] = (f16)rv;
    }
    if (tid < 256) {
      int b = tid >> 6, cc = tid & 63;
      float cv = enc[(b*CCH + cc)*N_ENC + (t / CREP)];
      A1[b*320 + 256 + cc] = (f16)cv;
      A3[b*192 + 128 + cc] = (f16)cv;
    }
    // prefetch layer-0 weights for this wave
    {
      const f16* nb1 = P1 + (size_t)w * (10*512) + lane*64;
      __builtin_prefetch(nb1, 0, 3);
      __builtin_prefetch(nb1 + 2048, 0, 3);
      __builtin_prefetch(nb1 + 4096, 0, 3);
      const f16* nb2 = P2 + (size_t)w * (4*512) + lane*64;
      __builtin_prefetch(nb2, 0, 3);
    }
    __syncthreads();
    if (tid < 4) xprevS[tid] = (float)sampleS[tid] * (1.f/127.5f) - 1.f;

    // ---- S2: skip = res @ Wskip0.T + bskip0 (waves 0..7) ----
    if (w < 8) {
      v8f acc = {0,0,0,0,0,0,0,0};
#pragma unroll
      for (int kt = 0; kt < 4; ++kt)
        acc = wmma16(ldsA(A2, 128, kt, lane), glbB(PS0, w*4 + kt, lane), acc);
      int n = w*16 + (lane & 15);
      float bias = bskip0[n];
      if (!(lane & 16)) {
#pragma unroll
        for (int v = 0; v < 4; ++v) skipS[v*128 + n] = acc[v] + bias;
      }
    }
    __syncthreads();

    // ---- dilated layers: 3 barriers per layer ----
    int qbase = 0;
    for (int l = 0; l < LAYERS; ++l) {
      const int d = 1 << (l & 7);
      // Phase B: z = A1 @ B1 + bc  (all 16 waves, K=320)
      {
        const f16* B = P1 + (size_t)l * P1_LAYER + (size_t)w * (10*512);
        v8f acc = {0,0,0,0,0,0,0,0};
#pragma unroll
        for (int kt = 0; kt < 10; ++kt)
          acc = wmma16(ldsA(A1, 320, kt, lane), glbB(B, kt, lane), acc);
        int n = w*16 + (lane & 15);
        int mb = (lane & 16) ? 8 : 0;
        float bias = bc[l*256 + n];
#pragma unroll
        for (int v = 0; v < 8; ++v)
          zbuf[(mb + v)*256 + n] = acc[v] + bias;
      }
      __syncthreads();
      // Phase C: h = sigmoid(gate)*tanh(out), rows 0..3; prefetch next weights
      {
        int m = tid >> 7, j = tid & 127;
        float g = zbuf[m*256 + j];
        float o = zbuf[m*256 + 128 + j];
        float h = (1.f / (1.f + __expf(-g))) * tanhf(o);
        A2[m*128 + j] = (f16)h;
        if (l < 15) {
          const f16* nb1 = P1 + (size_t)(l+1) * P1_LAYER + (size_t)w * (10*512) + lane*64;
          __builtin_prefetch(nb1, 0, 3);
          __builtin_prefetch(nb1 + 2048, 0, 3);
          __builtin_prefetch(nb1 + 4096, 0, 3);
          const f16* nb2 = P2 + (size_t)(l+1) * P2_LAYER + (size_t)w * (4*512) + lane*64;
          __builtin_prefetch(nb2, 0, 3);
        } else {
          if (w < 8) {                        // top fc weights
            const f16* nb3 = P3 + (size_t)w * (6*512) + lane*64;
            __builtin_prefetch(nb3, 0, 3);
            __builtin_prefetch(nb3 + 2048, 0, 3);
          } else {                            // logits weights
            const f16* nb4 = P4 + (size_t)(w - 8) * (2*4*512) + lane*64;
            __builtin_prefetch(nb4, 0, 3);
            __builtin_prefetch(nb4 + 2048, 0, 3);
          }
        }
      }
      __syncthreads();
      // Phase D: [skip_d | res_d] = h @ B2 ; owner-lane updates double as
      //          next-layer Phase A (queue r/w + A1 staging)
      if (w < 8) {                            // skip half (always needed)
        const f16* B = P2 + (size_t)l * P2_LAYER + (size_t)w * (4*512);
        v8f acc = {0,0,0,0,0,0,0,0};
#pragma unroll
        for (int kt = 0; kt < 4; ++kt)
          acc = wmma16(ldsA(A2, 128, kt, lane), glbB(B, kt, lane), acc);
        int n = w*16 + (lane & 15);
        float bias = bskip_l[l*128 + n];
        if (!(lane & 16)) {
#pragma unroll
          for (int v = 0; v < 4; ++v) {
            float ns = skipS[v*128 + n] + acc[v] + bias;
            skipS[v*128 + n] = ns;
            if (l == 15) A3[v*192 + n] = (f16)fmaxf(ns, 0.f);
          }
        }
      } else if (l < 15) {                    // res half (dead at last layer)
        const f16* B = P2 + (size_t)l * P2_LAYER + (size_t)w * (4*512);
        v8f acc = {0,0,0,0,0,0,0,0};
#pragma unroll
        for (int kt = 0; kt < 4; ++kt)
          acc = wmma16(ldsA(A2, 128, kt, lane), glbB(B, kt, lane), acc);
        int n = (w - 8)*16 + (lane & 15);
        float bias = bres[l*128 + n];
        const int dn    = 1 << ((l + 1) & 7);
        const int slotn = t & (dn - 1);
        float* Qn = Q + qbase + BATCH*RCH*d;  // next layer's ring buffer
        if (!(lane & 16)) {
#pragma unroll
          for (int v = 0; v < 4; ++v) {
            int idx = v*128 + n;
            float nr = resS[idx] + acc[v] + bias;
            resS[idx] = nr;
            float* qp = Qn + idx*dn + slotn;  // read t-dn, write current res
            float q0 = *qp;
            *qp = nr;
            A1[v*320 + n]       = (f16)q0;
            A1[v*320 + 128 + n] = (f16)nr;
          }
        }
      }
      __syncthreads();
      qbase += BATCH*RCH*d;
    }

    // ---- top: s = relu(skip) @ Wfc.T + bfc + c @ Wcond_top.T ----
    if (w < 8) {
      v8f acc = {0,0,0,0,0,0,0,0};
#pragma unroll
      for (int kt = 0; kt < 6; ++kt)
        acc = wmma16(ldsA(A3, 192, kt, lane), glbB(P3, w*6 + kt, lane), acc);
      int n = w*16 + (lane & 15);
      float bias = bfc[n];
      if (!(lane & 16)) {                     // owners write relu(s) A directly
#pragma unroll
        for (int v = 0; v < 4; ++v)
          A2[v*128 + n] = (f16)fmaxf(acc[v] + bias, 0.f);
      }
    }
    __syncthreads();
    // ---- logits = relu(s) @ Wlogits.T + blogits ----
    {
      const f16* B = P4 + (size_t)w * (4*512);
      v8f acc = {0,0,0,0,0,0,0,0};
#pragma unroll
      for (int kt = 0; kt < 4; ++kt)
        acc = wmma16(ldsA(A2, 128, kt, lane), glbB(B, kt, lane), acc);
      int n = w*16 + (lane & 15);
      if (!(lane & 16)) {                     // rows 0..3 live here
        float bias = blogits[n];
#pragma unroll
        for (int v = 0; v < 4; ++v) {
          float val = acc[v] + bias;
          logitsS[v*256 + n] = val;
          out[(size_t)v*(CLASSES*TSTEPS) + (size_t)n*TSTEPS + t] = val;
        }
      }
    }
    __syncthreads();
    // ---- argmax per batch row (waves 0..3, wave32 shuffle reduce) ----
    if (w < 4) {
      float bv = -3.4e38f; int bn = 0;
      for (int i = 0; i < 8; ++i) {           // ascending n per lane -> first-max
        int n = lane + 32*i;
        float v = logitsS[w*256 + n];
        if (v > bv) { bv = v; bn = n; }
      }
#pragma unroll
      for (int off = 16; off >= 1; off >>= 1) {
        float ov = __shfl_xor(bv, off, 32);
        int   on = __shfl_xor(bn, off, 32);
        if (ov > bv || (ov == bv && on < bn)) { bv = ov; bn = on; }
      }
      if (lane == 0) sampleS[w] = bn;
    }
    __syncthreads();
  }
}

extern "C" void kernel_launch(void* const* d_in, const int* in_sizes, int n_in,
                              void* d_out, int out_size, void* d_ws, size_t ws_size,
                              hipStream_t stream) {
  const float* enc       = (const float*)d_in[0];
  const float* W_first   = (const float*)d_in[1];
  const float* b_first   = (const float*)d_in[2];
  const float* Wc        = (const float*)d_in[3];
  const float* bc        = (const float*)d_in[4];
  const float* Wcond     = (const float*)d_in[5];
  const float* Wres      = (const float*)d_in[6];
  const float* bres      = (const float*)d_in[7];
  const float* Wskip_l   = (const float*)d_in[8];
  const float* bskip_l   = (const float*)d_in[9];
  const float* Wskip0    = (const float*)d_in[10];
  const float* bskip0    = (const float*)d_in[11];
  const float* Wcond_top = (const float*)d_in[12];
  const float* Wfc       = (const float*)d_in[13];
  const float* bfc       = (const float*)d_in[14];
  const float* Wlogits   = (const float*)d_in[15];
  const float* blogits   = (const float*)d_in[16];

  char* ws = (char*)d_ws;
  f16*   P1  = (f16*)(ws + WS_P1);
  f16*   P2  = (f16*)(ws + WS_P2);
  f16*   PS0 = (f16*)(ws + WS_PS0);
  f16*   P3  = (f16*)(ws + WS_P3);
  f16*   P4  = (f16*)(ws + WS_P4);
  float* Q   = (float*)(ws + WS_Q);
  float* out = (float*)d_out;

  wavenet_pack<<<(PACK_TOTAL + 255) / 256, 256, 0, stream>>>(
      Wc, Wcond, Wres, Wskip_l, Wskip0, Wcond_top, Wfc, Wlogits,
      P1, P2, PS0, P3, P4, Q);

  wavenet_gen<<<1, NTHREADS, 0, stream>>>(
      enc, W_first, b_first, bc, bres, bskip_l, bskip0, bfc, blogits,
      P1, P2, PS0, P3, P4, Q, out);
}
